// GNNCorrelationLearner_16174846837135
// MI455X (gfx1250) — compile-verified, matrix operand order checked by reading, not run on hardware
//
#include <hip/hip_runtime.h>
#include <math.h>

#define NNODES 50000
#define NEDGES 500000
#define F_NODE 7
#define F_EDGE 8
#define DD 64
#define HH 128
#define NLAYERS 3
#define KPAD 160          // 2D+Fe = 136 padded to 5 k-tiles of 32
#define MT 32             // edges per block-iteration (two 16-row WMMA M-tiles)

typedef __attribute__((ext_vector_type(16))) _Float16 v16h;
typedef __attribute__((ext_vector_type(8)))  float    v8f;

// ---------------------------------------------------------------------------
// CDNA5 16-bit WMMA fragment loaders (ISA 7.12.2 layouts, wave32)
// A (16x32, MxK) row-major with leading dim ld:
//   lanes 0-15: M=lane, K={0..7,16..23}; lanes 16-31: M=lane-16, K={8..15,24..31}
// Each fragment is two contiguous 16-byte runs per lane -> lowers to ds_load_b128.
// ---------------------------------------------------------------------------
__device__ inline v16h load_a_frag16(const _Float16* a, int ld, int lane) {
  const int m  = lane & 15;
  const int kh = (lane >> 4) * 8;          // 0 or 8
  const _Float16* r = a + m * ld;
  v16h f;
#pragma unroll
  for (int j = 0; j < 8; ++j) {
    const int k = ((j < 4) ? 0 : 16) + kh + (j & 3) * 2;
    f[2 * j]     = r[k];
    f[2 * j + 1] = r[k + 1];
  }
  return f;
}

// B (32x16, KxN) stored row-major (K-major) with leading dim ld:
//   lane -> column N, same per-lane K split as A.
__device__ inline v16h load_b_frag16(const _Float16* b, int ld, int lane) {
  const int n  = lane & 15;
  const int kh = (lane >> 4) * 8;
  v16h f;
#pragma unroll
  for (int j = 0; j < 8; ++j) {
    const int k = ((j < 4) ? 0 : 16) + kh + (j & 3) * 2;
    f[2 * j]     = b[k * ld + n];
    f[2 * j + 1] = b[(k + 1) * ld + n];
  }
  return f;
}

__device__ inline v8f wmma_f16(v16h a, v16h b, v8f c) {
  return __builtin_amdgcn_wmma_f32_16x16x32_f16(false, a, false, b,
                                                (short)0, c, false, false);
}

__device__ inline v8f bias_acc(float bias) {
  v8f c;
#pragma unroll
  for (int r = 0; r < 8; ++r) c[r] = bias;
  return c;
}

// ---------------------------------------------------------------------------
// Weight conversion/padding to f16 (one shot per launch)
// ---------------------------------------------------------------------------
__global__ void convert_weights_kernel(const float* __restrict__ msgW1,
                                       const float* __restrict__ msgW2,
                                       const float* __restrict__ predW1,
                                       const float* __restrict__ predW2,
                                       _Float16* __restrict__ wm1,
                                       _Float16* __restrict__ wm2,
                                       _Float16* __restrict__ wp1,
                                       _Float16* __restrict__ wp2) {
  int i = blockIdx.x * blockDim.x + threadIdx.x;
  const int n1 = NLAYERS * KPAD * HH;      // msg_W1 padded
  const int n2 = NLAYERS * HH * HH;        // msg_W2
  const int n3 = KPAD * HH;                // pred_W1 padded
  const int n4 = HH * DD;                  // pred_W2
  if (i < n1) {
    int l = i / (KPAD * HH), rem = i % (KPAD * HH);
    int k = rem / HH, n = rem % HH;
    float v = (k < 136) ? msgW1[(l * 136 + k) * HH + n] : 0.0f;
    wm1[i] = (_Float16)v;
    return;
  }
  i -= n1;
  if (i < n2) { wm2[i] = (_Float16)msgW2[i]; return; }
  i -= n2;
  if (i < n3) {
    int k = i / HH, n = i % HH;
    wp1[i] = (_Float16)((k < 136) ? predW1[k * HH + n] : 0.0f);
    return;
  }
  i -= n3;
  if (i < n4) wp2[i] = (_Float16)predW2[i];
}

// edge_attr -> f16 mirror (one shot)
__global__ __launch_bounds__(256) void convert_attr_kernel(const float* __restrict__ a,
                                                           _Float16* __restrict__ ah) {
  const size_t i = (size_t)blockIdx.x * 256 + threadIdx.x;
  if (i < (size_t)NEDGES * F_EDGE) ah[i] = (_Float16)a[i];
}

// ---------------------------------------------------------------------------
// Node embedding: x = relu(nf @ W + b); writes f32 master + f16 mirror
// ---------------------------------------------------------------------------
__global__ __launch_bounds__(128) void embed_kernel(const float* __restrict__ nf,
                                                    const float* __restrict__ W,
                                                    const float* __restrict__ b,
                                                    float* __restrict__ x,
                                                    _Float16* __restrict__ xh) {
  const int n = blockIdx.x * blockDim.x + threadIdx.x;
  if (n >= NNODES) return;
  float f[F_NODE];
#pragma unroll
  for (int i = 0; i < F_NODE; ++i) f[i] = nf[n * F_NODE + i];
  for (int j = 0; j < DD; ++j) {
    float acc = b[j];
#pragma unroll
    for (int i = 0; i < F_NODE; ++i) acc += f[i] * W[i * DD + j];
    const float v = acc > 0.0f ? acc : 0.0f;
    x[n * DD + j]  = v;
    xh[n * DD + j] = (_Float16)v;
  }
}

// ---------------------------------------------------------------------------
// In-degree counts (float, to match reference)
// ---------------------------------------------------------------------------
__global__ __launch_bounds__(256) void count_kernel(const int* __restrict__ dst,
                                                    float* __restrict__ counts) {
  const int e = blockIdx.x * blockDim.x + threadIdx.x;
  if (e < NEDGES) atomicAdd(&counts[dst[e]], 1.0f);
}

// ---------------------------------------------------------------------------
// Zero the 24 pad columns of the LDS A-tile (done once per block; the tile
// loop never writes cols 136..159 afterwards).
// ---------------------------------------------------------------------------
__device__ inline void zero_pad_tile(_Float16* sA, int tid) {
  const uint4 z = {0u, 0u, 0u, 0u};
  for (int i = tid; i < MT * 3; i += 256) {
    const int m = i / 3, u = i % 3;
    *(uint4*)(sA + m * KPAD + 136 + u * 8) = z;
  }
}

// ---------------------------------------------------------------------------
// A-tile build: pure f16 16-byte copies, no conversions.
// Per row: 8 chunks x[src], 8 chunks x[dst], 1 chunk edge_attr (17 total).
// ---------------------------------------------------------------------------
__device__ inline void build_edge_tile(_Float16* sA, const _Float16* __restrict__ xh,
                                       const _Float16* __restrict__ attrh,
                                       const int* sSrc, const int* sDst,
                                       int ebase, int tid) {
  for (int i = tid; i < MT * 17; i += 256) {
    const int m = i / 17, u = i % 17;
    const uint4* p;
    int dcol;
    if (u < 8)       { p = (const uint4*)(xh + (size_t)sSrc[m] * DD) + u;       dcol = u * 8; }
    else if (u < 16) { p = (const uint4*)(xh + (size_t)sDst[m] * DD) + (u - 8); dcol = u * 8; }
    else             { p = (const uint4*)(attrh + (size_t)(ebase + m) * F_EDGE); dcol = 128; }
    *(uint4*)(sA + m * KPAD + dcol) = *p;
  }
}

// ---------------------------------------------------------------------------
// Edge message MLP (hot loop): 32-edge tiles, bias folded into WMMA C,
// all A-fragments of a GEMM preloaded so ds loads pipeline ahead of WMMAs.
// ---------------------------------------------------------------------------
__global__ __launch_bounds__(256) void msg_kernel(const _Float16* __restrict__ xh,
                                                  const _Float16* __restrict__ attrh,
                                                  const int* __restrict__ src,
                                                  const int* __restrict__ dst,
                                                  const _Float16* __restrict__ w1,  // [KPAD][HH]
                                                  const float* __restrict__ b1,
                                                  const _Float16* __restrict__ w2,  // [HH][HH]
                                                  const float* __restrict__ b2,
                                                  float* __restrict__ agg) {
  __shared__ __align__(16) _Float16 sA[MT * KPAD];
  __shared__ __align__(16) _Float16 sH[MT * HH];
  __shared__ int sSrc[MT], sDst[MT];

  const int tid  = threadIdx.x;
  const int lane = tid & 31;
  const int wave = tid >> 5;               // 0..7
  const int ntiles = NEDGES / MT;

  v16h w1f[5], w2f[4];
#pragma unroll
  for (int kt = 0; kt < 5; ++kt)
    w1f[kt] = load_b_frag16(w1 + kt * 32 * HH + wave * 16, HH, lane);
#pragma unroll
  for (int kt = 0; kt < 4; ++kt)
    w2f[kt] = load_b_frag16(w2 + kt * 32 * HH + wave * 16, HH, lane);

  const int   ncol  = wave * 16 + (lane & 15);
  const float bias1 = b1[ncol];
  const float bias2 = b2[ncol];
  const int   mhalf = (lane >> 4) * 8;     // C/D: lanes>=16 hold M=8..15

  zero_pad_tile(sA, tid);

  for (int t = blockIdx.x; t < ntiles; t += gridDim.x) {
    __syncthreads();                       // protect sA/sH reuse across tiles
    if (tid < MT) { sSrc[tid] = src[t * MT + tid]; sDst[tid] = dst[t * MT + tid]; }
    if (tid == 0) {                        // prefetch next tile this block handles
      const int tn = t + gridDim.x;
      if (tn < ntiles) {
        __builtin_prefetch(src + tn * MT, 0, 0);
        __builtin_prefetch(dst + tn * MT, 0, 0);
        __builtin_prefetch(attrh + (size_t)tn * MT * F_EDGE, 0, 0);
      }
    }
    __syncthreads();

    build_edge_tile(sA, xh, attrh, sSrc, sDst, t * MT, tid);
    __syncthreads();

    // GEMM1: [32xKPAD] x [KPADx16] per wave (two M-tiles share B frags)
    v16h af[10];
#pragma unroll
    for (int kt = 0; kt < 5; ++kt) {
      af[kt]     = load_a_frag16(sA + kt * 32, KPAD, lane);
      af[5 + kt] = load_a_frag16(sA + 16 * KPAD + kt * 32, KPAD, lane);
    }
    v8f a0 = bias_acc(bias1), a1 = bias_acc(bias1);
#pragma unroll
    for (int kt = 0; kt < 5; ++kt) {
      a0 = wmma_f16(af[kt],     w1f[kt], a0);
      a1 = wmma_f16(af[5 + kt], w1f[kt], a1);
    }
#pragma unroll
    for (int r = 0; r < 8; ++r) {
      const float v0 = a0[r] > 0.0f ? a0[r] : 0.0f;
      const float v1 = a1[r] > 0.0f ? a1[r] : 0.0f;
      sH[(r + mhalf) * HH + ncol]      = (_Float16)v0;
      sH[(16 + r + mhalf) * HH + ncol] = (_Float16)v1;
    }
    __syncthreads();

    // GEMM2: [32x128] x [128x16] -> messages, scatter-add
    v16h hf[8];
#pragma unroll
    for (int kt = 0; kt < 4; ++kt) {
      hf[kt]     = load_a_frag16(sH + kt * 32, HH, lane);
      hf[4 + kt] = load_a_frag16(sH + 16 * HH + kt * 32, HH, lane);
    }
    v8f m0 = bias_acc(bias2), m1 = bias_acc(bias2);
#pragma unroll
    for (int kt = 0; kt < 4; ++kt) {
      m0 = wmma_f16(hf[kt],     w2f[kt], m0);
      m1 = wmma_f16(hf[4 + kt], w2f[kt], m1);
    }
#pragma unroll
    for (int r = 0; r < 8; ++r) {
      atomicAdd(&agg[(size_t)sDst[r + mhalf] * HH + ncol],      m0[r]);
      atomicAdd(&agg[(size_t)sDst[16 + r + mhalf] * HH + ncol], m1[r]);
    }
  }
}

// ---------------------------------------------------------------------------
// Node update + residual + layernorm (f32-exact); writes f32 + f16 mirror.
// ---------------------------------------------------------------------------
__global__ __launch_bounds__(64) void update_kernel(const float* __restrict__ xin,
                                                    const float* __restrict__ agg,
                                                    const float* __restrict__ counts,
                                                    const float* __restrict__ W,   // [192][64]
                                                    const float* __restrict__ b,
                                                    const float* __restrict__ g,
                                                    const float* __restrict__ beta,
                                                    float* __restrict__ xout,
                                                    _Float16* __restrict__ xhout) {
  __shared__ float sx[DD];
  __shared__ float sa[HH];
  __shared__ float red[DD];
  const int n = blockIdx.x;
  const int j = threadIdx.x;
  const float inv_cnt = 1.0f / (counts[n] + 1e-6f);
  sx[j] = xin[(size_t)n * DD + j];
  sa[j]      = agg[(size_t)n * HH + j] * inv_cnt;
  sa[j + 64] = agg[(size_t)n * HH + j + 64] * inv_cnt;
  __syncthreads();

  float acc = b[j];
  for (int k = 0; k < DD; ++k)  acc += sx[k] * W[k * DD + j];
  for (int k = 0; k < HH; ++k)  acc += sa[k] * W[(DD + k) * DD + j];
  const float v = sx[j] + (acc > 0.0f ? acc : 0.0f);

  red[j] = v;
  __syncthreads();
  for (int s = 32; s > 0; s >>= 1) { if (j < s) red[j] += red[j + s]; __syncthreads(); }
  const float mu = red[0] * (1.0f / DD);
  __syncthreads();
  const float dv = v - mu;
  red[j] = dv * dv;
  __syncthreads();
  for (int s = 32; s > 0; s >>= 1) { if (j < s) red[j] += red[j + s]; __syncthreads(); }
  const float var = red[0] * (1.0f / DD);
  const float y = dv * rsqrtf(var + 1e-5f) * g[j] + beta[j];
  xout[(size_t)n * DD + j]  = y;
  xhout[(size_t)n * DD + j] = (_Float16)y;
}

// ---------------------------------------------------------------------------
// Edge predictor: GEMM(KPADx128) relu GEMM(128x64) relu dot(64) tanh.
// GEMM2: 32x64 output = 8 wave-tiles -> one per wave (mt2=w>>2, nt2=w&3).
// ---------------------------------------------------------------------------
__global__ __launch_bounds__(256) void pred_kernel(const _Float16* __restrict__ xh,
                                                   const _Float16* __restrict__ attrh,
                                                   const int* __restrict__ src,
                                                   const int* __restrict__ dst,
                                                   const _Float16* __restrict__ w1,  // [KPAD][HH]
                                                   const float* __restrict__ b1,
                                                   const _Float16* __restrict__ w2,  // [HH][64]
                                                   const float* __restrict__ b2,
                                                   const float* __restrict__ w3,    // [64]
                                                   const float* __restrict__ b3,
                                                   float* __restrict__ out) {
  __shared__ __align__(16) _Float16 sA[MT * KPAD];
  __shared__ __align__(16) _Float16 sH[MT * HH];
  __shared__ float sH2[MT * DD];
  __shared__ int sSrc[MT], sDst[MT];

  const int tid  = threadIdx.x;
  const int lane = tid & 31;
  const int wave = tid >> 5;
  const int ntiles = NEDGES / MT;

  const int mt2 = wave >> 2;               // GEMM2 M-tile for this wave
  const int nt2 = wave & 3;                // GEMM2 N-tile for this wave

  v16h w1f[5], w2f[4];
#pragma unroll
  for (int kt = 0; kt < 5; ++kt)
    w1f[kt] = load_b_frag16(w1 + kt * 32 * HH + wave * 16, HH, lane);
#pragma unroll
  for (int kt = 0; kt < 4; ++kt)
    w2f[kt] = load_b_frag16(w2 + kt * 32 * DD + nt2 * 16, DD, lane);

  const int   ncol  = wave * 16 + (lane & 15);
  const int   ncol2 = nt2 * 16 + (lane & 15);
  const float bias1 = b1[ncol];
  const float bias2 = b2[ncol2];
  const float b3v   = b3[0];
  const int   mhalf = (lane >> 4) * 8;

  zero_pad_tile(sA, tid);

  for (int t = blockIdx.x; t < ntiles; t += gridDim.x) {
    __syncthreads();
    if (tid < MT) { sSrc[tid] = src[t * MT + tid]; sDst[tid] = dst[t * MT + tid]; }
    if (tid == 0) {
      const int tn = t + gridDim.x;
      if (tn < ntiles) {
        __builtin_prefetch(src + tn * MT, 0, 0);
        __builtin_prefetch(dst + tn * MT, 0, 0);
        __builtin_prefetch(attrh + (size_t)tn * MT * F_EDGE, 0, 0);
      }
    }
    __syncthreads();

    build_edge_tile(sA, xh, attrh, sSrc, sDst, t * MT, tid);
    __syncthreads();

    // GEMM1
    v16h af[10];
#pragma unroll
    for (int kt = 0; kt < 5; ++kt) {
      af[kt]     = load_a_frag16(sA + kt * 32, KPAD, lane);
      af[5 + kt] = load_a_frag16(sA + 16 * KPAD + kt * 32, KPAD, lane);
    }
    v8f a0 = bias_acc(bias1), a1 = bias_acc(bias1);
#pragma unroll
    for (int kt = 0; kt < 5; ++kt) {
      a0 = wmma_f16(af[kt],     w1f[kt], a0);
      a1 = wmma_f16(af[5 + kt], w1f[kt], a1);
    }
#pragma unroll
    for (int r = 0; r < 8; ++r) {
      const float v0 = a0[r] > 0.0f ? a0[r] : 0.0f;
      const float v1 = a1[r] > 0.0f ? a1[r] : 0.0f;
      sH[(r + mhalf) * HH + ncol]      = (_Float16)v0;
      sH[(16 + r + mhalf) * HH + ncol] = (_Float16)v1;
    }
    __syncthreads();

    // GEMM2: one 16x16 tile per wave
    v16h hf[4];
#pragma unroll
    for (int kt = 0; kt < 4; ++kt)
      hf[kt] = load_a_frag16(sH + mt2 * 16 * HH + kt * 32, HH, lane);
    v8f acc2 = bias_acc(bias2);
#pragma unroll
    for (int kt = 0; kt < 4; ++kt)
      acc2 = wmma_f16(hf[kt], w2f[kt], acc2);
#pragma unroll
    for (int r = 0; r < 8; ++r)
      sH2[(mt2 * 16 + r + mhalf) * DD + ncol2] = acc2[r] > 0.0f ? acc2[r] : 0.0f;
    __syncthreads();

    // Final 64->1 dot + tanh: one row per lane of wave 0
    if (tid < MT) {
      float acc3 = b3v;
      for (int k = 0; k < DD; ++k) acc3 += sH2[tid * DD + k] * w3[k];
      out[t * MT + tid] = tanhf(acc3);
    }
  }
}

// ---------------------------------------------------------------------------
// Host-side orchestration (graph-capture safe: only async ops on `stream`)
// ---------------------------------------------------------------------------
extern "C" void kernel_launch(void* const* d_in, const int* in_sizes, int n_in,
                              void* d_out, int out_size, void* d_ws, size_t ws_size,
                              hipStream_t stream) {
  const float* node_features = (const float*)d_in[0];
  const float* edge_attr     = (const float*)d_in[1];
  const float* embed_W       = (const float*)d_in[2];
  const float* embed_b       = (const float*)d_in[3];
  const float* msg_W1        = (const float*)d_in[4];
  const float* msg_b1        = (const float*)d_in[5];
  const float* msg_W2        = (const float*)d_in[6];
  const float* msg_b2        = (const float*)d_in[7];
  const float* upd_W         = (const float*)d_in[8];
  const float* upd_b         = (const float*)d_in[9];
  const float* ln_g          = (const float*)d_in[10];
  const float* ln_b          = (const float*)d_in[11];
  const float* pred_W1       = (const float*)d_in[12];
  const float* pred_b1       = (const float*)d_in[13];
  const float* pred_W2       = (const float*)d_in[14];
  const float* pred_b2       = (const float*)d_in[15];
  const float* pred_W3       = (const float*)d_in[16];
  const float* pred_b3       = (const float*)d_in[17];
  const int*   edge_index    = (const int*)d_in[18];
  const int* src = edge_index;
  const int* dst = edge_index + NEDGES;
  float* out = (float*)d_out;

  char* ws = (char*)d_ws;
  size_t off = 0;
  auto take = [&](size_t bytes) -> void* {
    void* p = ws + off;
    off += (bytes + 255) & ~(size_t)255;
    return p;
  };
  float*     xA     = (float*)take((size_t)NNODES * DD * 4);
  float*     xB     = (float*)take((size_t)NNODES * DD * 4);
  float*     agg    = (float*)take((size_t)NNODES * HH * 4);
  float*     counts = (float*)take((size_t)NNODES * 4);
  _Float16*  xhA    = (_Float16*)take((size_t)NNODES * DD * 2);
  _Float16*  xhB    = (_Float16*)take((size_t)NNODES * DD * 2);
  _Float16*  attrh  = (_Float16*)take((size_t)NEDGES * F_EDGE * 2);
  _Float16*  wm1    = (_Float16*)take((size_t)NLAYERS * KPAD * HH * 2);
  _Float16*  wm2    = (_Float16*)take((size_t)NLAYERS * HH * HH * 2);
  _Float16*  wp1    = (_Float16*)take((size_t)KPAD * HH * 2);
  _Float16*  wp2    = (_Float16*)take((size_t)HH * DD * 2);

  // 1) convert weights + edge attrs to f16
  {
    const int total = NLAYERS * KPAD * HH + NLAYERS * HH * HH + KPAD * HH + HH * DD;
    convert_weights_kernel<<<(total + 255) / 256, 256, 0, stream>>>(
        msg_W1, msg_W2, pred_W1, pred_W2, wm1, wm2, wp1, wp2);
    const size_t na = (size_t)NEDGES * F_EDGE;
    convert_attr_kernel<<<(int)((na + 255) / 256), 256, 0, stream>>>(edge_attr, attrh);
  }

  // 2) node embedding (f32 + f16 mirror)
  embed_kernel<<<(NNODES + 127) / 128, 128, 0, stream>>>(node_features, embed_W, embed_b,
                                                         xA, xhA);

  // 3) in-degree counts
  hipMemsetAsync(counts, 0, (size_t)NNODES * 4, stream);
  count_kernel<<<(NEDGES + 255) / 256, 256, 0, stream>>>(dst, counts);

  // 4) message-passing layers
  float*    xcur = xA;  float*    xnxt = xB;
  _Float16* hcur = xhA; _Float16* hnxt = xhB;
  for (int l = 0; l < NLAYERS; ++l) {
    hipMemsetAsync(agg, 0, (size_t)NNODES * HH * 4, stream);
    msg_kernel<<<2048, 256, 0, stream>>>(
        hcur, attrh, src, dst,
        wm1 + (size_t)l * KPAD * HH, msg_b1 + l * HH,
        wm2 + (size_t)l * HH * HH,   msg_b2 + l * HH, agg);
    update_kernel<<<NNODES, 64, 0, stream>>>(
        xcur, agg, counts,
        upd_W + (size_t)l * (DD + HH) * DD, upd_b + l * DD,
        ln_g + l * DD, ln_b + l * DD, xnxt, hnxt);
    float* tf = xcur; xcur = xnxt; xnxt = tf;
    _Float16* th = hcur; hcur = hnxt; hnxt = th;
  }

  // 5) edge predictor
  pred_kernel<<<2048, 256, 0, stream>>>(
      hcur, attrh, src, dst,
      wp1, pred_b1, wp2, pred_b2, pred_W3, pred_b3, out);
}
